// SNN_Model_59227599012529
// MI455X (gfx1250) — matrix operand order repaired and checked by reading, not compiled, not run on hardware
//
#include <hip/hip_runtime.h>
#include <hip/hip_bf16.h>

// ---------------------------------------------------------------------------
// Persistent SNN forward for MI455X (gfx1250, wave32, WMMA).
// 16 batch rows per workgroup, 128 threads (4 waves), 100 time steps in-kernel.
// W2 lives in LDS in pre-swizzled WMMA B-fragment layout (247 KB) -- enabled
// by CDNA5's 320 KB LDS per workgroup. The GEMM inner loop uses IGroupLP
// sched_group_barrier directives to interleave next-tile LDS fragment loads
// between the current tile's WMMAs (hides LDS latency at 1 wave/SIMD).
// ---------------------------------------------------------------------------

typedef __attribute__((ext_vector_type(16))) _Float16 v16h;
typedef __attribute__((ext_vector_type(8)))  _Float16 v8h;
typedef __attribute__((ext_vector_type(8)))  float    v8f;

#define THREADS 128
#define TILE_M  16
#define D_IN    128
#define H1      400
#define H2      300
#define O_OUT   3
#define KT2     13     // ceil(400/32): K padded to 416 with zeros
#define NT2     19     // ceil(300/16): N padded to 304 with zeros
#define KT3     10     // ceil(304/32): layer-3 K padded to 320 with zeros
#define BETA    0.95f

#if __has_builtin(__builtin_amdgcn_sched_group_barrier)
#define SCHED_GROUP(mask, size) __builtin_amdgcn_sched_group_barrier((mask), (size), 0)
#else
#define SCHED_GROUP(mask, size)
#endif
#define SG_WMMA    0x008   // MFMA/WMMA group
#define SG_DSREAD  0x100   // DS read group

struct __align__(1024) Smem {
  _Float16 a1[KT2 * 512];            // spk1 A-fragments (also x staging)  13,312 B
  _Float16 w2[NT2 * KT2 * 512];      // W2 B-fragments (f16)              252,928 B
  _Float16 cur1[TILE_M * H1];        // constant input current             12,800 B
  float    mem1[TILE_M * H1];        // layer-1 membrane                   25,600 B
  _Float16 a3[KT3 * 512];            // spk2 A-fragments                   10,240 B
  _Float16 w3[KT3 * 512];            // W3 B-fragments                     10,240 B
};                                   // total = 325,120 B  (<= 320 KiB LDS)

// Half-index inside a 1 KB A-fragment for element (row m, k-in-tile kk), per
// CDNA5 ISA 16-bit A-matrix 16x32 layout (lanes 0-15: K 0-7,16-23;
// lanes 16-31: K 8-15,24-31).
__device__ __forceinline__ int afrag_idx(int m, int kk) {
  int lane = m + ((kk & 8) ? 16 : 0);
  int half = (kk & 7) + ((kk & 16) ? 8 : 0);
  return lane * 16 + half;           // lane stride = 16 halves (32 B)
}

__device__ __forceinline__ v8f v8f_zero() {
  v8f r;
#pragma unroll
  for (int i = 0; i < 8; ++i) r[i] = 0.f;
  return r;
}

__global__ __launch_bounds__(THREADS)
void snn_persistent_kernel(const float* __restrict__ x,
                           const float* __restrict__ W1,
                           const float* __restrict__ b1,
                           const float* __restrict__ W2,
                           const float* __restrict__ b2,
                           const float* __restrict__ W3,
                           const float* __restrict__ b3,
                           const int*   __restrict__ nsp,
                           float*       __restrict__ out) {
  __shared__ Smem s;
  const int tid  = threadIdx.x;
  const int wave = tid >> 5;
  const int lane = tid & 31;
  const int row0 = blockIdx.x * TILE_M;
  const int nsteps = nsp[0];

  // ---- stage x tile (f32) into a1 region ------------------------------------
  float* xl = (float*)s.a1;
  for (int i = tid; i < TILE_M * D_IN; i += THREADS)
    xl[i] = x[(row0 + (i >> 7)) * D_IN + (i & 127)];
  __syncthreads();

  // ---- cur1 = x @ W1^T + b1  (one-time, f32 math, stored f16) ---------------
  for (int i = tid; i < TILE_M * H1; i += THREADS) {
    int m = i / H1, n = i - m * H1;
    const float4* wr = (const float4*)(W1 + n * D_IN);
    const float4* xr = (const float4*)(xl + m * D_IN);
    float acc = b1[n];
#pragma unroll 8
    for (int k4 = 0; k4 < D_IN / 4; ++k4) {
      float4 w = wr[k4], xv = xr[k4];
      acc += w.x * xv.x + w.y * xv.y + w.z * xv.z + w.w * xv.w;
    }
    s.cur1[i] = (_Float16)acc;
  }
  __syncthreads();

  // ---- zero fragment regions + membrane; pack W2/W3 into B-frag layout ------
  for (int i = tid; i < KT2 * 512; i += THREADS) s.a1[i] = (_Float16)0.f;
  for (int i = tid; i < KT3 * 512; i += THREADS) s.a3[i] = (_Float16)0.f;
  for (int i = tid; i < TILE_M * H1; i += THREADS) s.mem1[i] = 0.f;
  // B-frag layout: lane l<16 holds K=kt*32+{0-7,16-23} for column N=nt*16+l;
  //                lane l>=16 holds K=kt*32+{8-15,24-31} for column N=nt*16+(l-16).
  for (int e = tid; e < NT2 * KT2 * 512; e += THREADS) {
    int nt = e / (KT2 * 512);
    int r  = e - nt * (KT2 * 512);
    int kt = r >> 9;
    int q  = r & 511;
    int ln = q >> 4, hf = q & 15;
    int k = kt * 32 + ((ln & 16) ? 8 : 0) + (hf & 7) + ((hf & 8) ? 16 : 0);
    int n = nt * 16 + (ln & 15);
    _Float16 v = (_Float16)0.f;
    if (k < H1 && n < H2) v = (_Float16)W2[n * H1 + k];
    s.w2[e] = v;
  }
  for (int e = tid; e < KT3 * 512; e += THREADS) {
    int kt = e >> 9;
    int q  = e & 511;
    int ln = q >> 4, hf = q & 15;
    int k = kt * 32 + ((ln & 16) ? 8 : 0) + (hf & 7) + ((hf & 8) ? 16 : 0);
    int n = ln & 15;
    _Float16 v = (_Float16)0.f;
    if (k < H2 && n < O_OUT) v = (_Float16)W3[n * H2 + k];
    s.w3[e] = v;
  }

  // ---- per-wave persistent state -------------------------------------------
  float b2v[5];
#pragma unroll
  for (int sl = 0; sl < 5; ++sl) {
    int nt = wave + 4 * sl;
    int n  = nt * 16 + (lane & 15);
    b2v[sl] = (nt < NT2 && n < H2) ? b2[n] : 0.f;
  }
  float b3v = ((lane & 15) < O_OUT) ? b3[lane & 15] : 0.f;

  v8f mem2[5];
#pragma unroll
  for (int sl = 0; sl < 5; ++sl) mem2[sl] = v8f_zero();
  v8f mem3 = v8f_zero();
  v8f accO = v8f_zero();

  __syncthreads();

  // ======================= time loop (all 100 steps) ========================
  for (int t = 0; t < nsteps; ++t) {
    // ---- layer-1 LIF, fragment-row-major: each A-frag row (kt, fl) holds two
    //      runs of 8 consecutive k for a fixed m -> all-vector LDS traffic ----
    for (int r = tid; r < KT2 * 32; r += THREADS) {   // 416 fragment rows
      int kt = r >> 5, fl = r & 31;
      int m  = fl & 15;
      int k0 = kt * 32 + ((fl & 16) ? 8 : 0);
      int base = m * H1 + k0;
      v16h spk;
      {  // run 0: k0..k0+7 (always < 400)
        v8h c  = *(const v8h*)(s.cur1 + base);
        v8f mm = *(const v8f*)(s.mem1 + base);
#pragma unroll
        for (int j = 0; j < 8; ++j) {
          float mv    = mm[j];
          float reset = (mv > 1.0f) ? 1.0f : 0.0f;        // spike(mem - 1)
          mv = BETA * mv + (float)c[j] - reset;           // reset-by-subtract
          mm[j]  = mv;
          spk[j] = (mv > 1.0f) ? (_Float16)1.0f : (_Float16)0.0f;
        }
        *(v8f*)(s.mem1 + base) = mm;
      }
      if (kt < KT2 - 1) {  // run 1: k0+16..k0+23 (k >= 400 only when kt == 12)
        v8h c  = *(const v8h*)(s.cur1 + base + 16);
        v8f mm = *(const v8f*)(s.mem1 + base + 16);
#pragma unroll
        for (int j = 0; j < 8; ++j) {
          float mv    = mm[j];
          float reset = (mv > 1.0f) ? 1.0f : 0.0f;
          mv = BETA * mv + (float)c[j] - reset;
          mm[j]      = mv;
          spk[8 + j] = (mv > 1.0f) ? (_Float16)1.0f : (_Float16)0.0f;
        }
        *(v8f*)(s.mem1 + base + 16) = mm;
      } else {
#pragma unroll
        for (int j = 0; j < 8; ++j) spk[8 + j] = (_Float16)0.0f;
      }
      *(v16h*)(s.a1 + kt * 512 + fl * 16) = spk;
    }
    __syncthreads();

    // ---- layer-2 GEMM: cur2 = spk1 @ W2^T, 2-deep software pipeline --------
    v8f acc[5];
#pragma unroll
    for (int sl = 0; sl < 5; ++sl) acc[sl] = v8f_zero();

    const _Float16* a1p = s.a1 + lane * 16;
    const _Float16* w2p[5];
#pragma unroll
    for (int sl = 0; sl < 5; ++sl) {
      int nt  = wave + 4 * sl;
      int nte = (nt < NT2) ? nt : 0;          // dummy tile keeps EXEC full
      w2p[sl] = s.w2 + nte * (KT2 * 512) + lane * 16;
    }

    v16h Ac, An, Bc[5], Bn[5];
    Ac = *(const v16h*)(a1p);
#pragma unroll
    for (int sl = 0; sl < 5; ++sl) Bc[sl] = *(const v16h*)(w2p[sl]);
    SCHED_GROUP(SG_DSREAD, 12);               // prologue: A(0) + 5x B(0)

#pragma unroll
    for (int kt = 0; kt < KT2; ++kt) {
      if (kt + 1 < KT2) {                     // prefetch next k-tile
        An = *(const v16h*)(a1p + (kt + 1) * 512);
#pragma unroll
        for (int sl = 0; sl < 5; ++sl)
          Bn[sl] = *(const v16h*)(w2p[sl] + (kt + 1) * 512);
      }
#pragma unroll
      for (int sl = 0; sl < 5; ++sl)          // 5 independent accumulators
        acc[sl] = __builtin_amdgcn_wmma_f32_16x16x32_f16(
            false, Ac, false, Bc[sl], (short)0, acc[sl], false, false);

      // IGroupLP: slot next-tile fragment loads between this tile's WMMAs
      if (kt + 1 < KT2) {
#pragma unroll
        for (int g = 0; g < 5; ++g) {
          SCHED_GROUP(SG_WMMA, 1);            // 1 WMMA ...
          SCHED_GROUP(SG_DSREAD, 2);          // ... then 2 LDS b128 reads
        }
        SCHED_GROUP(SG_DSREAD, 2);            // next A-fragment reads
        Ac = An;
#pragma unroll
        for (int sl = 0; sl < 5; ++sl) Bc[sl] = Bn[sl];
      } else {
        SCHED_GROUP(SG_WMMA, 5);              // last tile: WMMAs only
      }
    }

    // ---- layer-2 LIF on C-fragment registers; scatter spk2 into A3 frags ---
#pragma unroll
    for (int sl = 0; sl < 5; ++sl) {
      int nt = wave + 4 * sl;
      if (nt < NT2) {
#pragma unroll
        for (int v = 0; v < 8; ++v) {
          float cur   = acc[sl][v] + b2v[sl];
          float mm    = mem2[sl][v];
          float reset = (mm > 1.0f) ? 1.0f : 0.0f;
          mm = BETA * mm + cur - reset;
          mem2[sl][v] = mm;
          int mrow = v + ((lane >= 16) ? 8 : 0);         // C-frag row
          int kcol = nt * 16 + (lane & 15);              // h2 index -> layer-3 K
          s.a3[(kcol >> 5) * 512 + afrag_idx(mrow, kcol & 31)] =
              (mm > 1.0f) ? (_Float16)1.0f : (_Float16)0.0f;
        }
      }
    }
    __syncthreads();

    // ---- layer-3: one 16x16 tile on wave 0 (K padded to 320), pipelined ----
    if (wave == 0) {
      v8f acc3 = v8f_zero();
      v16h a3c = *(const v16h*)(s.a3 + lane * 16);
      v16h w3c = *(const v16h*)(s.w3 + lane * 16);
      SCHED_GROUP(SG_DSREAD, 4);
#pragma unroll
      for (int kt = 0; kt < KT3; ++kt) {
        v16h a3n, w3n;
        if (kt + 1 < KT3) {
          a3n = *(const v16h*)(s.a3 + (kt + 1) * 512 + lane * 16);
          w3n = *(const v16h*)(s.w3 + (kt + 1) * 512 + lane * 16);
        }
        acc3 = __builtin_amdgcn_wmma_f32_16x16x32_f16(
            false, a3c, false, w3c, (short)0, acc3, false, false);
        if (kt + 1 < KT3) {
          SCHED_GROUP(SG_WMMA, 1);
          SCHED_GROUP(SG_DSREAD, 4);
          a3c = a3n; w3c = w3n;
        } else {
          SCHED_GROUP(SG_WMMA, 1);
        }
      }
#pragma unroll
      for (int v = 0; v < 8; ++v) {
        float cur   = acc3[v] + b3v;
        float mm    = mem3[v];
        float reset = (mm > 1.0f) ? 1.0f : 0.0f;
        mm = BETA * mm + cur - reset;
        mem3[v] = mm;
        accO[v] += (mm > 1.0f) ? 1.0f : 0.0f;
      }
    }
    // a1 rewrites next iteration are fenced by the post-matmul barrier above;
    // a3 rewrites happen only after the post-layer-1 barrier, which wave 0
    // reaches only after finishing its a3 reads.
  }

  // ---- output: sigmoid(acc / num_steps), wave 0 holds the C-fragment -------
  if (wave == 0) {
    int n = lane & 15;
    if (n < O_OUT) {
#pragma unroll
      for (int v = 0; v < 8; ++v) {
        int row = row0 + v + ((lane >= 16) ? 8 : 0);
        float val = accO[v] / (float)nsteps;
        out[row * O_OUT + n] = 1.0f / (1.0f + __expf(-val));
      }
    }
  }
}

extern "C" void kernel_launch(void* const* d_in, const int* in_sizes, int n_in,
                              void* d_out, int out_size, void* d_ws, size_t ws_size,
                              hipStream_t stream) {
  (void)n_in; (void)out_size; (void)d_ws; (void)ws_size;
  const float* x  = (const float*)d_in[0];
  const float* W1 = (const float*)d_in[1];
  const float* b1 = (const float*)d_in[2];
  const float* W2 = (const float*)d_in[3];
  const float* b2 = (const float*)d_in[4];
  const float* W3 = (const float*)d_in[5];
  const float* b3 = (const float*)d_in[6];
  const int*   ns = (const int*)d_in[7];
  float* out = (float*)d_out;

  int B    = in_sizes[0] / D_IN;   // 4096
  int grid = B / TILE_M;           // 256 workgroups, 16 rows each
  snn_persistent_kernel<<<dim3(grid), dim3(THREADS), 0, stream>>>(
      x, W1, b1, W2, b2, W3, b3, ns, out);
}
